// StackedBLSTMEmbedding_79139067396606
// MI455X (gfx1250) — matrix operand order repaired and checked by reading, not compiled, not run on hardware
//
#include <hip/hip_runtime.h>

typedef __attribute__((ext_vector_type(16))) __bf16         v16bf;
typedef __attribute__((ext_vector_type(16))) unsigned short v16u;
typedef __attribute__((ext_vector_type(8)))  float          v8f;
typedef __attribute__((ext_vector_type(4)))  unsigned int   u32x4;
typedef __attribute__((ext_vector_type(8)))  int            i32x8;
typedef __attribute__((ext_vector_type(4)))  int            i32x4;

#if defined(__has_builtin)
#if __has_builtin(__builtin_amdgcn_tensor_load_to_lds)
#define HAVE_TDM 1
#else
#define HAVE_TDM 0
#endif
#else
#define HAVE_TDM 0
#endif

#define WMMA_BF16(a, b, c) \
    __builtin_amdgcn_wmma_f32_16x16x32_bf16(false, (a), false, (b), (short)0, (c), false, false)

__device__ __forceinline__ unsigned short f32_to_bf16(float f) {
    unsigned int u = __float_as_uint(f);
    u += 0x7FFFu + ((u >> 16) & 1u);          // round-to-nearest-even
    return (unsigned short)(u >> 16);
}

__device__ __forceinline__ v16bf ldfrag(const unsigned short* p) {
    v16u u = *(const v16u*)p;
    return __builtin_bit_cast(v16bf, u);
}

__device__ __forceinline__ float sigm(float x) { return 1.0f / (1.0f + __expf(-x)); }

#if HAVE_TDM
// 1-D TDM copy of 64 KB (8192 x 8B elements) from global to LDS, per CDNA5 D#
// group layout (CDNA5 ISA). 6-arg builtin
// (amdgpu-toolchain / therock-10.0): (u32x4, i32x8, i32x4, i32x4, i32x8, cpol).
__device__ __forceinline__ void tdm_copy_64k(const float* gsrc, unsigned lds_off) {
    unsigned long long ga = (unsigned long long)gsrc;
    u32x4 g0;
    g0[0] = 1u;                                            // count=1 (valid user D#)
    g0[1] = lds_off;                                       // lds_addr (bytes)
    g0[2] = (unsigned)(ga & 0xFFFFFFFFu);                  // global_addr[31:0]
    g0[3] = (unsigned)((ga >> 32) & 0x1FFFFFFu) | (2u << 30); // addr[56:32] | type=2
    i32x8 g1;
    g1[0] = (3 << 16);            // workgroup_mask=0, data_size=3 (8B), no pad/iter
    g1[1] = (8192 << 16);         // tensor_dim0[15:0]=8192 in [31:16]
    g1[2] = (1 << 16);            // tensor_dim0 hi=0, tensor_dim1=1
    g1[3] = (8192 << 16);         // tensor_dim1 hi=0, tile_dim0=8192
    g1[4] = 1;                    // tile_dim1=1, tile_dim2=0
    g1[5] = 8192;                 // tensor_dim0_stride lo
    g1[6] = 0;                    // stride0 hi, tensor_dim1_stride lo16
    g1[7] = 0;
    i32x4 g2; g2[0] = 0; g2[1] = 0; g2[2] = 0; g2[3] = 0;
    i32x4 g3; g3[0] = 0; g3[1] = 0; g3[2] = 0; g3[3] = 0;
    i32x8 g4;
    g4[0] = 0; g4[1] = 0; g4[2] = 0; g4[3] = 0;
    g4[4] = 0; g4[5] = 0; g4[6] = 0; g4[7] = 0;
    __builtin_amdgcn_tensor_load_to_lds(g0, g1, g2, g3, g4, 0);
}
#endif

// ---------------------------------------------------------------------------
// Whh (1024x256 f32, row = gate col N) -> bf16 WMMA B-fragment layout:
//   out[((g*8 + kt)*32 + lane)*16 + idx], n = g*16+(lane&15),
//   k = kt*32 + (idx>>3)*16 + (lane>>4)*8 + (idx&7)
// ---------------------------------------------------------------------------
__global__ __launch_bounds__(256) void convert_whh_kernel(
    const float* __restrict__ w1f, const float* __restrict__ w1b,
    const float* __restrict__ w2f, const float* __restrict__ w2b,
    unsigned short* __restrict__ out)
{
    int id = blockIdx.x * 256 + threadIdx.x;           // 4 * 262144 total
    int mat  = id >> 18;
    int rem  = id & 262143;
    int idx  = rem & 15;
    int lane = (rem >> 4) & 31;
    int kt   = (rem >> 9) & 7;
    int g    = rem >> 12;
    int n    = g * 16 + (lane & 15);
    int k    = kt * 32 + ((idx >> 3) << 4) + ((lane >> 4) << 3) + (idx & 7);
    const float* src = (mat == 0) ? w1f : (mat == 1) ? w1b : (mat == 2) ? w2f : w2b;
    out[id] = f32_to_bf16(src[n * 256 + k]);
}

// ---------------------------------------------------------------------------
// C = A * W^T + bias.  aTrT != 0: logical A-row r maps to physical row
// (r%32)*aTrT + r/32  (i.e. A is (B,T,*) but we iterate rows in (t,b) order).
// cFrag != 0: store C in WMMA C-fragment tile layout
//   C[(((row/16)*(N/16) + col/16)*32 + lane)*8 + r]  (contiguous per row-tile).
// ---------------------------------------------------------------------------
__global__ __launch_bounds__(256) void gemm_bias_kernel(
    const float* __restrict__ A, const float* __restrict__ W,
    const float* __restrict__ bias, float* __restrict__ C,
    int M, int N, int K, int aTrT, int cFrag)
{
    __shared__ __align__(32) unsigned short sA[4][32][16];   // A-frag layout
    __shared__ __align__(32) unsigned short sB[4][32][16];   // B-frag layout

    const int tid  = threadIdx.x;
    const int w    = tid >> 5;
    const int lane = tid & 31;
    const int mtile = w & 3;
    const int nt0   = (w >> 2) << 1;
    const int bm = blockIdx.y * 64, bn = blockIdx.x * 64;
    const int nn = lane & 15;

    float b0v = bias[bn + nt0 * 16 + nn];
    float b1v = bias[bn + (nt0 + 1) * 16 + nn];
    v8f acc0, acc1;
#pragma unroll
    for (int r = 0; r < 8; ++r) { acc0[r] = b0v; acc1[r] = b1v; }

    const int nk = (K + 31) >> 5;
    for (int kc = 0; kc < nk; ++kc) {
        const int k0 = kc << 5;
#pragma unroll
        for (int e = tid; e < 1024; e += 256) {        // pairs of k
            int m = e >> 4;
            int k = (e & 15) << 1;
            int gk = k0 + k;
            float a0 = 0.f, a1 = 0.f, bv0 = 0.f, bv1 = 0.f;
            if (gk < K) {                              // K even -> whole pair
                int grow = bm + m;
                int arow = aTrT ? ((grow & 31) * aTrT + (grow >> 5)) : grow;
                float2 av = *(const float2*)(A + (size_t)arow * K + gk);
                float2 bv = *(const float2*)(W + (size_t)(bn + m) * K + gk);
                a0 = av.x; a1 = av.y; bv0 = bv.x; bv1 = bv.y;
            }
            unsigned pa = (unsigned)f32_to_bf16(a0) | ((unsigned)f32_to_bf16(a1) << 16);
            unsigned pb = (unsigned)f32_to_bf16(bv0) | ((unsigned)f32_to_bf16(bv1) << 16);
            int wi = k & 15;
            *(unsigned*)&sA[m >> 4][(m & 15) + ((wi & 8) ? 16 : 0)][(k >> 4) * 8 + (wi & 7)] = pa;
            *(unsigned*)&sB[m >> 4][(m & 15) + ((wi & 8) ? 16 : 0)][(k >> 4) * 8 + (wi & 7)] = pb;
        }
        __syncthreads();
        v16bf a  = ldfrag(&sA[mtile][lane][0]);
        v16bf f0 = ldfrag(&sB[nt0][lane][0]);
        v16bf f1 = ldfrag(&sB[nt0 + 1][lane][0]);
        acc0 = WMMA_BF16(a, f0, acc0);
        acc1 = WMMA_BF16(a, f1, acc1);
        __syncthreads();
    }
    if (cFrag) {
        const size_t nt16 = (size_t)(N >> 4);
        const size_t rt   = (size_t)(bm >> 4) + mtile;
        *(v8f*)(C + ((rt * nt16 + (bn >> 4) + nt0)     * 32 + lane) * 8) = acc0;
        *(v8f*)(C + ((rt * nt16 + (bn >> 4) + nt0 + 1) * 32 + lane) * 8) = acc1;
    } else {
#pragma unroll
        for (int r = 0; r < 8; ++r) {
            int row = bm + mtile * 16 + r + 8 * (lane >> 4);
            C[(size_t)row * N + bn + nt0 * 16 + nn]       = acc0[r];
            C[(size_t)row * N + bn + (nt0 + 1) * 16 + nn] = acc1[r];
        }
    }
}

// ---------------------------------------------------------------------------
// Persistent LSTM recurrence. Grid = 4 blocks: (dir, batch-half); 16 waves,
// wave w owns hidden j-tile w for all four gates. h in LDS (A-frag bf16),
// c in VGPRs. xg is in C-fragment layout with rows ordered (t*32+b), so one
// timestep of one batch-half is a contiguous 64 KB block: forward blocks
// double-buffer it into LDS with the Tensor Data Mover; backward blocks
// gather per-row reversed indices directly (+ prefetch of the next step).
// ---------------------------------------------------------------------------
__global__ __launch_bounds__(512) void lstm_kernel(
    const float* __restrict__ xgF, const float* __restrict__ xgR,
    const unsigned short* __restrict__ whhF, const unsigned short* __restrict__ whhR,
    float* __restrict__ y, const int* __restrict__ x_len,
    int len_div, int T)
{
    __shared__ __align__(32) unsigned short sH[8][32][16];   // h, A-frag layout
#if HAVE_TDM
    extern __shared__ float xbuf[];                          // 2 x 16384 floats
#endif
    const int tid  = threadIdx.x;
    const int dir  = blockIdx.x >> 1;                  // 0 fwd, 1 bwd
    const int bb   = (blockIdx.x & 1) * 16;            // batch base
    const int bbh  = bb >> 4;
    const int w    = tid >> 5;                         // j-tile 0..15
    const int lane = tid & 31;
    const int lh   = lane >> 4;
    const int nn   = lane & 15;

    const float* xg          = dir ? xgR  : xgF;
    const unsigned short* wh = dir ? whhR : whhF;

    for (int i = tid; i < 8 * 32 * 16; i += 512) ((unsigned short*)sH)[i] = 0;

    int lenv[8];
#pragma unroll
    for (int r = 0; r < 8; ++r) lenv[r] = x_len[bb + r + 8 * lh] / len_div;
    float c[8];
#pragma unroll
    for (int r = 0; r < 8; ++r) c[r] = 0.0f;

    const int j = w * 16 + nn;                         // hidden col 0..255
    const int kt_out  = j >> 5;
    const int kk      = j & 31;
    const int wi      = kk & 15;
    const int ln_base = (wi & 8) ? 16 : 0;
    const int idx_st  = (kk >> 4) * 8 + (wi & 7);

#if HAVE_TDM
    const bool ftdm = (dir == 0);
    const unsigned dynoff = (unsigned)__builtin_amdgcn_groupstaticsize();
    if (ftdm && w == 0)
        tdm_copy_64k(xg + (size_t)bbh * 16384, dynoff);   // tile for s=0 -> buf0
#endif
    __syncthreads();

    for (int s = 0; s < T; ++s) {
        v8f aI, aF, aG, aO;
#if HAVE_TDM
        if (ftdm) {
            int cur = s & 1;
            if (w == 0) {
                if (s + 1 < T) {                       // overlap next step's DMA
                    tdm_copy_64k(xg + (size_t)((s + 1) * 2 + bbh) * 16384,
                                 dynoff + (unsigned)(cur ^ 1) * 65536u);
                    __builtin_amdgcn_s_wait_tensorcnt(1);
                } else {
                    __builtin_amdgcn_s_wait_tensorcnt(0);
                }
            }
            __syncthreads();                           // buf[cur] ready for all
            const float* bp = xbuf + cur * 16384;
            aI = *(const v8f*)(bp + ((0 * 16 + w) * 32 + lane) * 8);
            aF = *(const v8f*)(bp + ((1 * 16 + w) * 32 + lane) * 8);
            aG = *(const v8f*)(bp + ((2 * 16 + w) * 32 + lane) * 8);
            aO = *(const v8f*)(bp + ((3 * 16 + w) * 32 + lane) * 8);
        } else
#endif
        {
#pragma unroll
            for (int r = 0; r < 8; ++r) {
                int t = dir ? (lenv[r] - 1 - s) : s;
                if (t < 0) t = 0;                      // matches jnp.clip
                const float* p0 = xg + (((size_t)(t * 2 + bbh) * 64 + w) * 32 + lane) * 8 + r;
                aI[r] = p0[0 * 4096];
                aF[r] = p0[1 * 4096];
                aG[r] = p0[2 * 4096];
                aO[r] = p0[3 * 4096];
            }
            {   // prefetch next step's slice (speculative, clamped in-buffer)
                int tn = dir ? (lenv[0] - 2 - s) : (s + 1);
                if (tn < 0) tn = 0;
                if (tn >= T) tn = T - 1;
                const float* p0 = xg + (((size_t)(tn * 2 + bbh) * 64 + w) * 32 + lane) * 8;
                __builtin_prefetch(p0, 0, 1);
                __builtin_prefetch(p0 + 4096, 0, 1);
                __builtin_prefetch(p0 + 8192, 0, 1);
                __builtin_prefetch(p0 + 12288, 0, 1);
            }
        }
        // G += h @ Whh^T for this wave's four gate tiles
#pragma unroll
        for (int kt = 0; kt < 8; ++kt) {
            v16bf a  = ldfrag(&sH[kt][lane][0]);
            v16bf bI = ldfrag(wh + (((size_t)(0 * 16 + w) * 8 + kt) * 32 + lane) * 16);
            v16bf bF = ldfrag(wh + (((size_t)(1 * 16 + w) * 8 + kt) * 32 + lane) * 16);
            v16bf bG = ldfrag(wh + (((size_t)(2 * 16 + w) * 8 + kt) * 32 + lane) * 16);
            v16bf bO = ldfrag(wh + (((size_t)(3 * 16 + w) * 8 + kt) * 32 + lane) * 16);
            aI = WMMA_BF16(a, bI, aI);
            aF = WMMA_BF16(a, bF, aF);
            aG = WMMA_BF16(a, bG, aG);
            aO = WMMA_BF16(a, bO, aO);
        }
        __syncthreads();                               // all done reading old h
#pragma unroll
        for (int r = 0; r < 8; ++r) {
            float cn = sigm(aF[r]) * c[r] + sigm(aI[r]) * tanhf(aG[r]);
            float hn = sigm(aO[r]) * tanhf(cn);
            if (s < lenv[r]) {                         // mask: update + emit
                c[r] = cn;
                int b     = bb + r + 8 * lh;
                int t_out = dir ? (lenv[r] - 1 - s) : s;
                y[(size_t)(b * T + t_out) * 512 + dir * 256 + j] = hn;
                sH[kt_out][(r + 8 * lh) + ln_base][idx_st] = f32_to_bf16(hn);
            }
        }
        __syncthreads();                               // new h visible everywhere
    }
}

// ---------------------------------------------------------------------------
__global__ __launch_bounds__(256) void pool_kernel(
    const float* __restrict__ y, float* __restrict__ p)
{
    int id = blockIdx.x * 256 + threadIdx.x;           // 16,384,000 total
    int cc = id & 511;
    int to = (id >> 9) % 1000;
    int b  = id / (512 * 1000);
    const float* base = y + ((size_t)b * 2000 + 2 * to) * 512 + cc;
    p[id] = fmaxf(base[0], base[512]);
}

__global__ void lens_kernel(const int* __restrict__ x_len, int* __restrict__ out) {
    int b = threadIdx.x;
    if (b < 32) out[b] = x_len[b] / 2;
}

// ---------------------------------------------------------------------------
extern "C" void kernel_launch(void* const* d_in, const int* in_sizes, int n_in,
                              void* d_out, int out_size, void* d_ws, size_t ws_size,
                              hipStream_t stream) {
    (void)in_sizes; (void)n_in; (void)out_size; (void)ws_size;
    const float* x     = (const float*)d_in[0];
    const int*   x_len = (const int*)d_in[1];
    const float* Wih1f = (const float*)d_in[2];
    const float* Whh1f = (const float*)d_in[3];
    const float* b1f   = (const float*)d_in[4];
    const float* Wih1b = (const float*)d_in[5];
    const float* Whh1b = (const float*)d_in[6];
    const float* b1b   = (const float*)d_in[7];
    const float* Wih2f = (const float*)d_in[8];
    const float* Whh2f = (const float*)d_in[9];
    const float* b2f   = (const float*)d_in[10];
    const float* Wih2b = (const float*)d_in[11];
    const float* Whh2b = (const float*)d_in[12];
    const float* b2b   = (const float*)d_in[13];
    const float* Wlin  = (const float*)d_in[14];
    const float* blin  = (const float*)d_in[15];
    float* out = (float*)d_out;

    char* ws = (char*)d_ws;
    size_t off = 0;
    unsigned short* whhfrag = (unsigned short*)(ws + off); off += 4ull * 262144 * 2;
    float* xgA = (float*)(ws + off); off += 262144000ull;   // frag layout, (t,b) rows
    float* xgB = (float*)(ws + off); off += 262144000ull;
    float* y1  = (float*)(ws + off); off += 131072000ull;   // (32,2000,512) row-major
    float* p1  = (float*)(ws + off); off += 65536000ull;    // (32,1000,512) row-major

    convert_whh_kernel<<<4096, 256, 0, stream>>>(Whh1f, Whh1b, Whh2f, Whh2b, whhfrag);

    // Layer-1 input projections, rows in (t,b) order, C-fragment output layout
    gemm_bias_kernel<<<dim3(16, 1000), 256, 0, stream>>>(x, Wih1f, b1f, xgA,
                                                         64000, 1024, 80, 2000, 1);
    gemm_bias_kernel<<<dim3(16, 1000), 256, 0, stream>>>(x, Wih1b, b1b, xgB,
                                                         64000, 1024, 80, 2000, 1);

    (void)hipMemsetAsync(y1, 0, 131072000ull, stream);
    lstm_kernel<<<4, 512, 131072, stream>>>(xgA, xgB,
                                            whhfrag + 0 * 262144, whhfrag + 1 * 262144,
                                            y1, x_len, 1, 2000);

    pool_kernel<<<64000, 256, 0, stream>>>(y1, p1);

    // Layer-2 input projections
    gemm_bias_kernel<<<dim3(16, 500), 256, 0, stream>>>(p1, Wih2f, b2f, xgA,
                                                        32000, 1024, 512, 1000, 1);
    gemm_bias_kernel<<<dim3(16, 500), 256, 0, stream>>>(p1, Wih2b, b2b, xgB,
                                                        32000, 1024, 512, 1000, 1);

    (void)hipMemsetAsync(y1, 0, 65536000ull, stream);  // reuse y1 as y2
    lstm_kernel<<<4, 512, 131072, stream>>>(xgA, xgB,
                                            whhfrag + 2 * 262144, whhfrag + 3 * 262144,
                                            y1, x_len, 2, 1000);

    // Final linear -> d_out (row-major)
    gemm_bias_kernel<<<dim3(8, 500), 256, 0, stream>>>(y1, Wlin, blin, out,
                                                       32000, 512, 512, 0, 0);

    lens_kernel<<<1, 32, 0, stream>>>(x_len, (int*)d_out + 16384000);
}
